// SDIMLayer_57724360458322
// MI455X (gfx1250) — matrix operand order repaired and checked by reading, not compiled, not run on hardware
//
#include <hip/hip_runtime.h>
#include <hip/hip_bf16.h>
#include <math.h>

typedef __attribute__((ext_vector_type(16))) _Float16 v16h;
typedef __attribute__((ext_vector_type(8)))  _Float16 v8h;
typedef __attribute__((ext_vector_type(8)))  float    v8f;

#define B_   1024
#define L_   512
#define S_   64
#define EMB_ 16
#define E_   48
#define G_   4
#define C_   8
#define NH_  8
#define KD_  4
#define HD_  32      // NH_*KD_
#define FEAT_ 224    // 80 + 48 + 48 + 48, == 7*32 (exact WMMA K multiple)
#define N1_  200
#define N2_  80
#define KC_  7       // K chunks of 32 (K = 224)
#define NT1_ 13      // n-tiles layer 1 (208 cols, 200 valid)
#define NT2_ 5       // n-tiles layer 2 (80 cols)

// ---------------------------------------------------------------------------
// Kernel 0: pack fp32 weights into the exact per-lane WMMA B-fragment layout
// (ISA §7.12.2: lane l holds col n = nt*16+(l&15), K = kc*32+(e>>3)*16+(l>>4)*8+(e&7)).
// Packed element i = ((nt*KC_+kc)*32 + lane)*16 + e  -> each lane's fragment is
// 32 contiguous bytes => two clause-able global_load_b128 in the GEMM.
// ---------------------------------------------------------------------------
__global__ void pack_b(const float* __restrict__ src, _Float16* __restrict__ dst,
                       int Kvalid, int Nvalid, int NT)
{
    int i = blockIdx.x * blockDim.x + threadIdx.x;
    int total = NT * KC_ * 32 * 16;
    if (i >= total) return;
    int e     = i & 15;
    int lane  = (i >> 4) & 31;
    int chunk = i >> 9;                 // nt*KC_ + kc
    int kc = chunk % KC_, nt = chunk / KC_;
    int n = nt * 16 + (lane & 15);
    int k = kc * 32 + (e >> 3) * 16 + (lane >> 4) * 8 + (e & 7);
    float v = (k < Kvalid && n < Nvalid) ? src[k * Nvalid + n] : 0.0f;
    dst[i] = (_Float16)v;
}

// ---------------------------------------------------------------------------
// Kernel 1: one workgroup per batch row. Fuses embedding gather, LSH bucket
// scatter-mean (LDS float atomics), long-term interest gather, target
// attention, and assembles the 224-d feature row as f16 for the WMMA MLP.
// ---------------------------------------------------------------------------
__global__ __launch_bounds__(256) void feat_kernel(
    const int* __restrict__ uid, const int* __restrict__ ut1,
    const int* __restrict__ ut2, const int* __restrict__ ut3,
    const int* __restrict__ ut4,
    const int* __restrict__ lbg, const int* __restrict__ lbs, const int* __restrict__ lbc,
    const int* __restrict__ ltg, const int* __restrict__ lts, const int* __restrict__ ltc,
    const int* __restrict__ stg, const int* __restrict__ sts, const int* __restrict__ stc,
    const float* __restrict__ table, const float* __restrict__ Hw,
    const float* __restrict__ wq, const float* __restrict__ bq,
    const float* __restrict__ wk, const float* __restrict__ bk,
    const float* __restrict__ wv, const float* __restrict__ bv,
    const float* __restrict__ wo, const float* __restrict__ bo,
    _Float16* __restrict__ featH)
{
    const int b = blockIdx.x, tid = threadIdx.x;

    __shared__ float sH[E_ * G_ * 3];        // LSH projections (48,4,3)
    __shared__ float sXi[E_];                // item embedding
    __shared__ float sFeat[FEAT_];           // assembled feature row
    __shared__ float sCsum[G_][C_][E_];      // bucket sums
    __shared__ float sCnt[G_][C_];           // bucket counts (incl. pads, per reference)
    __shared__ float sXs[S_][E_];            // short-term embeddings
    __shared__ float sKm[S_][HD_];
    __shared__ float sVm[S_][HD_];
    __shared__ float sQ[HD_];
    __shared__ float sScore[NH_][S_];
    __shared__ float sCtx[HD_];
    __shared__ int   sItemCode[G_];
    __shared__ int   sLen;

    // ---- phase 1: load H, item/user embeddings, zero accumulators ----
    for (int i = tid; i < E_ * G_ * 3; i += 256) sH[i] = Hw[i];
    for (int j = tid; j < E_; j += 256) {
        int f = j >> 4, d = j & 15;
        int id = (f == 0 ? lbg[b] : (f == 1 ? lbs[b] : lbc[b]));
        float v = table[id * EMB_ + d];
        sXi[j] = v; sFeat[80 + j] = v;
    }
    for (int j = tid; j < 80; j += 256) {
        int f = j >> 4, d = j & 15;
        const int* p = (f == 0 ? uid : (f == 1 ? ut1 : (f == 2 ? ut2 : (f == 3 ? ut3 : ut4))));
        sFeat[j] = table[p[b] * EMB_ + d];
    }
    for (int i = tid; i < G_ * C_ * E_; i += 256) ((float*)sCsum)[i] = 0.0f;
    for (int i = tid; i < G_ * C_; i += 256) ((float*)sCnt)[i] = 0.0f;
    if (tid == 0) sLen = 0;
    __syncthreads();

    // ---- phase 2: item LSH codes + long-term bucket scatter ----
    if (tid < G_) {
        int g = tid; float d0 = 0, d1 = 0, d2 = 0;
        #pragma unroll
        for (int e = 0; e < E_; e++) {
            int base = e * 12 + g * 3; float x = sXi[e];
            d0 += x * sH[base]; d1 += x * sH[base + 1]; d2 += x * sH[base + 2];
        }
        sItemCode[g] = (d0 > 0.f ? 1 : 0) | (d1 > 0.f ? 2 : 0) | (d2 > 0.f ? 4 : 0);
    }
    for (int t = tid; t < L_; t += 256) {
        int i0 = ltg[b * L_ + t], i1 = lts[b * L_ + t], i2 = ltc[b * L_ + t];
        bool pad = (i0 == 0);
        float x[E_];
        #pragma unroll
        for (int d = 0; d < EMB_; d++) {
            x[d]      = table[i0 * EMB_ + d];
            x[16 + d] = table[i1 * EMB_ + d];
            x[32 + d] = table[i2 * EMB_ + d];
        }
        #pragma unroll
        for (int g = 0; g < G_; g++) {
            float d0 = 0, d1 = 0, d2 = 0;
            #pragma unroll
            for (int e = 0; e < E_; e++) {
                int base = e * 12 + g * 3; float xe = x[e];
                d0 += xe * sH[base]; d1 += xe * sH[base + 1]; d2 += xe * sH[base + 2];
            }
            int code = (d0 > 0.f ? 1 : 0) | (d1 > 0.f ? 2 : 0) | (d2 > 0.f ? 4 : 0);
            atomicAdd(&sCnt[g][code], 1.0f);            // counts include pads (reference vc)
            if (!pad) {
                #pragma unroll
                for (int e = 0; e < E_; e++) atomicAdd(&sCsum[g][code][e], x[e]);
            }
        }
    }
    __syncthreads();

    // ---- phase 3: long-term interest, short-term embedding load, seq length ----
    if (tid < E_) {
        float acc = 0.0f;
        #pragma unroll
        for (int g = 0; g < G_; g++) {
            int c = sItemCode[g]; float n = sCnt[g][c];
            if (n > 0.f) acc += sCsum[g][c][tid] / n;
        }
        sFeat[128 + tid] = acc * 0.25f;
    }
    for (int j = tid; j < S_ * E_; j += 256) {
        int k = j / E_, e = j - k * E_, f = e >> 4, d = e & 15;
        const int* p = (f == 0 ? stg : (f == 1 ? sts : stc));
        sXs[k][e] = table[p[b * S_ + k] * EMB_ + d];
    }
    if (tid < S_) { if (stg[b * S_ + tid] != 0) atomicAdd(&sLen, 1); }
    __syncthreads();

    // ---- phase 4: K, V, Q projections ----
    for (int idx = tid; idx < 2 * S_ * HD_; idx += 256) {
        bool isK = idx < S_ * HD_;
        int r = idx & (S_ * HD_ - 1);
        int k = r >> 5, hd = r & 31;
        const float* w  = isK ? wk : wv;
        const float* bb = isK ? bk : bv;
        float acc = bb[hd];
        #pragma unroll
        for (int e = 0; e < E_; e++) acc += sXs[k][e] * w[e * HD_ + hd];
        if (isK) sKm[k][hd] = acc; else sVm[k][hd] = acc;
    }
    if (tid < HD_) {
        float acc = bq[tid];
        #pragma unroll
        for (int e = 0; e < E_; e++) acc += sXi[e] * wq[e * HD_ + tid];
        sQ[tid] = acc;
    }
    __syncthreads();

    // ---- phase 5: masked scores ----
    for (int idx = tid; idx < NH_ * S_; idx += 256) {
        int h = idx >> 6, k = idx & (S_ - 1);
        float s = 0.0f;
        #pragma unroll
        for (int d = 0; d < KD_; d++) s += sQ[h * KD_ + d] * sKm[k][h * KD_ + d];
        s *= 0.5f;                          // 1/sqrt(KD)
        if (k >= sLen) s -= 1e9f;
        sScore[h][k] = s;
    }
    __syncthreads();

    // ---- phase 6: softmax per head ----
    if (tid < NH_) {
        int h = tid; float m = -INFINITY;
        for (int k = 0; k < S_; k++) m = fmaxf(m, sScore[h][k]);
        float sum = 0.0f;
        for (int k = 0; k < S_; k++) { float e = expf(sScore[h][k] - m); sScore[h][k] = e; sum += e; }
        float inv = 1.0f / sum;
        for (int k = 0; k < S_; k++) sScore[h][k] *= inv;
    }
    __syncthreads();

    // ---- phase 7: context ----
    if (tid < HD_) {
        int h = tid >> 2; float acc = 0.0f;
        for (int k = 0; k < S_; k++) acc += sScore[h][k] * sVm[k][tid];
        sCtx[tid] = acc;
    }
    __syncthreads();

    // ---- phase 8: output projection ----
    if (tid < E_) {
        float acc = bo[tid];
        #pragma unroll
        for (int hd = 0; hd < HD_; hd++) acc += sCtx[hd] * wo[hd * E_ + tid];
        sFeat[176 + tid] = acc;
    }
    __syncthreads();

    for (int j = tid; j < FEAT_; j += 256)
        featH[(size_t)b * FEAT_ + j] = (_Float16)sFeat[j];
}

// ---------------------------------------------------------------------------
// Kernels 2/4: WMMA GEMM, fully unrolled K (7 chunks of 32), f16 in / f32 out.
// A: row-major (M x 224), per-lane 2 contiguous b128 loads per chunk.
// B: pre-packed fragments, per-lane 32 contiguous bytes per chunk.
// Bias folded into accumulator init (bias is per-column; one column per lane).
// 4 waves per block, one 16x16 m-tile per wave sharing the n-tile's B.
// ---------------------------------------------------------------------------
__global__ __launch_bounds__(128) void wmma_gemm(
    const _Float16* __restrict__ A, const _Float16* __restrict__ Bp,
    const float* __restrict__ bias, float* __restrict__ C, int N)
{
    const int wave  = threadIdx.x >> 5;
    const int lane  = threadIdx.x & 31;
    const int mBase = (blockIdx.x * 4 + wave) * 16;
    const int nt    = blockIdx.y;
    const int half  = lane >> 4;
    const int l16   = lane & 15;
    const int col   = nt * 16 + l16;

    float bv = (col < N) ? bias[col] : 0.0f;
    v8f acc = { bv, bv, bv, bv, bv, bv, bv, bv };

    const _Float16* Ar = A + (size_t)(mBase + l16) * FEAT_;
    const _Float16* Bl = Bp + ((size_t)(nt * KC_) * 32 + lane) * 16;

    #pragma unroll
    for (int kc = 0; kc < KC_; ++kc) {
        v8h alo = *(const v8h*)(Ar + kc * 32 + half * 8);        // K = half*8 + 0..7
        v8h ahi = *(const v8h*)(Ar + kc * 32 + 16 + half * 8);   // K = 16 + half*8 + 0..7
        v16h bf = *(const v16h*)(Bl + (size_t)kc * 512);          // packed fragment
        v16h a;
        #pragma unroll
        for (int i = 0; i < 8; ++i) { a[i] = alo[i]; a[8 + i] = ahi[i]; }
        acc = __builtin_amdgcn_wmma_f32_16x16x32_f16(
                  false, a, false, bf, (short)0, acc, false, false);
    }
    if (col < N) {
        #pragma unroll
        for (int r = 0; r < 8; ++r)              // C/D: VGPR r -> row 8*half + r
            C[(size_t)(mBase + half * 8 + r) * N + col] = acc[r];
    }
}

// ---------------------------------------------------------------------------
// Kernel 3: LayerNorm(200) + ReLU (bias already in GEMM), emit f16 row
// padded to 224 with zeros for the next WMMA layer.
// ---------------------------------------------------------------------------
__global__ __launch_bounds__(256) void ln_relu1(
    const float* __restrict__ C1,
    const float* __restrict__ g1, const float* __restrict__ be1,
    _Float16* __restrict__ H1h)
{
    const int row = blockIdx.x, tid = threadIdx.x;
    __shared__ float red[256];
    float x = 0.0f;
    if (tid < N1_) x = C1[(size_t)row * N1_ + tid];
    red[tid] = (tid < N1_) ? x : 0.0f;
    __syncthreads();
    for (int s = 128; s > 0; s >>= 1) { if (tid < s) red[tid] += red[tid + s]; __syncthreads(); }
    float mu = red[0] * (1.0f / N1_);
    __syncthreads();
    float dd = x - mu;
    red[tid] = (tid < N1_) ? dd * dd : 0.0f;
    __syncthreads();
    for (int s = 128; s > 0; s >>= 1) { if (tid < s) red[tid] += red[tid + s]; __syncthreads(); }
    float inv = rsqrtf(red[0] * (1.0f / N1_) + 1e-3f);
    float y = 0.0f;
    if (tid < N1_) y = fmaxf(g1[tid] * dd * inv + be1[tid], 0.0f);
    if (tid < FEAT_) H1h[(size_t)row * FEAT_ + tid] = (_Float16)y;
}

// ---------------------------------------------------------------------------
// Kernel 5: LayerNorm(80) + ReLU + dot(w3) + sigmoid -> out[row]
// ---------------------------------------------------------------------------
__global__ __launch_bounds__(128) void ln2_final(
    const float* __restrict__ C2,
    const float* __restrict__ g2, const float* __restrict__ be2,
    const float* __restrict__ w3, const float* __restrict__ b3,
    float* __restrict__ out)
{
    const int row = blockIdx.x, tid = threadIdx.x;
    __shared__ float red[128];
    float x = 0.0f;
    if (tid < N2_) x = C2[(size_t)row * N2_ + tid];
    red[tid] = x;
    __syncthreads();
    for (int s = 64; s > 0; s >>= 1) { if (tid < s) red[tid] += red[tid + s]; __syncthreads(); }
    float mu = red[0] * (1.0f / N2_);
    __syncthreads();
    float dd = x - mu;
    red[tid] = (tid < N2_) ? dd * dd : 0.0f;
    __syncthreads();
    for (int s = 64; s > 0; s >>= 1) { if (tid < s) red[tid] += red[tid + s]; __syncthreads(); }
    float inv = rsqrtf(red[0] * (1.0f / N2_) + 1e-3f);
    __syncthreads();
    float y = 0.0f;
    if (tid < N2_) y = fmaxf(g2[tid] * dd * inv + be2[tid], 0.0f);
    red[tid] = (tid < N2_) ? y * w3[tid] : 0.0f;
    __syncthreads();
    for (int s = 64; s > 0; s >>= 1) { if (tid < s) red[tid] += red[tid + s]; __syncthreads(); }
    if (tid == 0) out[row] = 1.0f / (1.0f + expf(-(red[0] + b3[0])));
}

// ---------------------------------------------------------------------------
extern "C" void kernel_launch(void* const* d_in, const int* in_sizes, int n_in,
                              void* d_out, int out_size, void* d_ws, size_t ws_size,
                              hipStream_t stream) {
    (void)in_sizes; (void)n_in; (void)out_size; (void)ws_size;
    const int* uid = (const int*)d_in[0];
    const int* ut1 = (const int*)d_in[1];
    const int* ut2 = (const int*)d_in[2];
    const int* ut3 = (const int*)d_in[3];
    const int* ut4 = (const int*)d_in[4];
    const int* lbg = (const int*)d_in[5];
    const int* lbs = (const int*)d_in[6];
    const int* lbc = (const int*)d_in[7];
    const int* ltg = (const int*)d_in[8];
    const int* lts = (const int*)d_in[9];
    const int* ltc = (const int*)d_in[10];
    const int* stg = (const int*)d_in[11];
    const int* sts = (const int*)d_in[12];
    const int* stc = (const int*)d_in[13];
    const float* table = (const float*)d_in[14];
    const float* Hw    = (const float*)d_in[15];
    const float* wq = (const float*)d_in[16];  const float* bq = (const float*)d_in[17];
    const float* wk = (const float*)d_in[18];  const float* bk = (const float*)d_in[19];
    const float* wv = (const float*)d_in[20];  const float* bv = (const float*)d_in[21];
    const float* wo = (const float*)d_in[22];  const float* bo = (const float*)d_in[23];
    const float* w1 = (const float*)d_in[24];  const float* b1 = (const float*)d_in[25];
    const float* g1 = (const float*)d_in[26];  const float* be1 = (const float*)d_in[27];
    const float* w2 = (const float*)d_in[28];  const float* b2 = (const float*)d_in[29];
    const float* g2 = (const float*)d_in[30];  const float* be2 = (const float*)d_in[31];
    const float* w3 = (const float*)d_in[32];  const float* b3 = (const float*)d_in[33];

    // workspace layout (256B aligned slices)
    char* base = (char*)d_ws;
    size_t off = 0;
    auto take = [&](size_t bytes) -> char* {
        char* p = base + off;
        off = (off + bytes + 255) & ~(size_t)255;
        return p;
    };
    _Float16* featH = (_Float16*)take((size_t)B_ * FEAT_ * sizeof(_Float16));
    _Float16* w1p   = (_Float16*)take((size_t)NT1_ * KC_ * 512 * sizeof(_Float16));
    _Float16* w2p   = (_Float16*)take((size_t)NT2_ * KC_ * 512 * sizeof(_Float16));
    float*    C1    = (float*)   take((size_t)B_ * N1_ * sizeof(float));
    _Float16* H1h   = (_Float16*)take((size_t)B_ * FEAT_ * sizeof(_Float16));
    float*    C2    = (float*)   take((size_t)B_ * N2_ * sizeof(float));

    pack_b<<<(NT1_ * KC_ * 512 + 255) / 256, 256, 0, stream>>>(w1, w1p, FEAT_, N1_, NT1_);
    pack_b<<<(NT2_ * KC_ * 512 + 255) / 256, 256, 0, stream>>>(w2, w2p, N1_,   N2_, NT2_);

    feat_kernel<<<B_, 256, 0, stream>>>(
        uid, ut1, ut2, ut3, ut4, lbg, lbs, lbc,
        ltg, lts, ltc, stg, sts, stc,
        table, Hw, wq, bq, wk, bk, wv, bv, wo, bo, featH);

    wmma_gemm<<<dim3(B_ / 64, NT1_), 128, 0, stream>>>(featH, w1p, b1, C1, N1_);
    ln_relu1<<<B_, 256, 0, stream>>>(C1, g1, be1, H1h);
    wmma_gemm<<<dim3(B_ / 64, NT2_), 128, 0, stream>>>(H1h, w2p, b2, C2, N2_);
    ln2_final<<<B_, 128, 0, stream>>>(C2, g2, be2, w3, b3, (float*)d_out);
}
